// liGRU_29996051595635
// MI455X (gfx1250) — matrix operand order repaired and checked by reading, not compiled, not run on hardware
//
#include <hip/hip_runtime.h>

// ---------------------------------------------------------------------------
// Li-GRU (2 layers) for MI455X / gfx1250, wave32, bf16 WMMA w/ f32 accumulate.
// ---------------------------------------------------------------------------

typedef __attribute__((ext_vector_type(16))) __bf16 v16bf;
typedef __attribute__((ext_vector_type(8)))  __bf16 v8bf;
typedef __attribute__((ext_vector_type(8)))  float  v8f;

#define T_STEPS  1500
#define BATCH    16
#define FEAT     440
#define FEAT_PAD 448          // 14 * 32
#define HID      512
#define ROWS     (T_STEPS * BATCH)   // 24000

// Join two 16-byte bf16 chunks into one v16bf WMMA operand register group.
__device__ __forceinline__ v16bf join16(const __bf16* p0, const __bf16* p1) {
    v8bf lo = *(const v8bf*)p0;
    v8bf hi = *(const v8bf*)p1;
    v16bf r;
#pragma unroll
    for (int i = 0; i < 8; ++i) { r[i] = lo[i]; r[i + 8] = hi[i]; }
    return r;
}

__device__ __forceinline__ v8f wmma_bf16(v16bf a, v16bf b, v8f c) {
    return __builtin_amdgcn_wmma_f32_16x16x32_bf16(false, a, false, b,
                                                   (short)0, c, false, false);
}

// ---------------------------------------------------------------------------
// fp32 -> bf16 conversion with optional column zero-padding (K 440 -> 448).
// ---------------------------------------------------------------------------
__global__ void cvt_pad(const float* __restrict__ src, __bf16* __restrict__ dst,
                        int rows, int sc, int dc) {
    int idx = blockIdx.x * blockDim.x + threadIdx.x;
    if (idx >= rows * dc) return;
    int r = idx / dc, c = idx - r * dc;
    dst[idx] = (c < sc) ? (__bf16)src[r * sc + c] : (__bf16)0.0f;
}

// ---------------------------------------------------------------------------
// Fused gate projection: Ch = A x Wh^T + bh ; Cz = A x Wz^T + bz  (bf16 out).
// A[24000,K], W[512,K] row-major. One 16x16 output tile (both gates) per wave;
// the A operand is loaded once per k-step and feeds two WMMAs.
// grid = (ROWS/16, 512/16/8) = (1500, 4), block = 256 (8 waves).
// ---------------------------------------------------------------------------
__global__ void __launch_bounds__(256)
proj_gemm2(const __bf16* __restrict__ A,
           const __bf16* __restrict__ Wh, const __bf16* __restrict__ Wz,
           const float* __restrict__ bh,  const float* __restrict__ bz,
           __bf16* __restrict__ Ch, __bf16* __restrict__ Cz, int K) {
    const int lane  = threadIdx.x & 31;
    const int wave  = threadIdx.x >> 5;
    const int m0    = blockIdx.x * 16;
    const int n0    = (blockIdx.y * 8 + wave) * 16;

    // A operand: lane l -> row m = l&15; lanes<16 hold K 0..15, lanes>=16 hold K 16..31
    const int arow   = m0 + (lane & 15);
    const int khalfA = (lane >> 4) * 16;
    // B operand: lane l -> W row n; lanes<16 hold K 0..15, lanes>=16 hold K 16..31
    const int brow    = n0 + (lane & 15);
    const int kstartB = (lane >> 4) * 16;

    const __bf16* aP  = A  + (long)arow * K + khalfA;
    const __bf16* bhP = Wh + (long)brow * K + kstartB;
    const __bf16* bzP = Wz + (long)brow * K + kstartB;

    v8f acch = {};
    v8f accz = {};
    for (int k0 = 0; k0 < K; k0 += 32) {
        v16bf a  = join16(aP  + k0, aP  + k0 + 8);
        v16bf wh = join16(bhP + k0, bhP + k0 + 8);
        v16bf wz = join16(bzP + k0, bzP + k0 + 8);
        acch = wmma_bf16(a, wh, acch);
        accz = wmma_bf16(a, wz, accz);
    }

    const int   n     = n0 + (lane & 15);
    const float bhv   = bh[n];
    const float bzv   = bz[n];
    const int   mbase = (lane >> 4) * 8;   // C layout: VGPR r -> M=r (lo lanes) / r+8 (hi)
#pragma unroll
    for (int r = 0; r < 8; ++r) {
        long o = (long)(m0 + mbase + r) * HID + n;
        Ch[o] = (__bf16)(acch[r] + bhv);
        Cz[o] = (__bf16)(accz[r] + bzv);
    }
}

// ---------------------------------------------------------------------------
// Recurrence: one persistent 32-wave workgroup. h in LDS (fp32 + bf16 copy),
// Uh/Uz stream from L2. Each wave owns a 16-wide output column tile and both
// gates. K is split into two independent halves per gate (4 accumulators) to
// cut the dependent-WMMA chain from 16 to 8 and keep the matrix pipe fed.
// Next timestep's gate tiles are prefetched during the WMMA loop.
// ---------------------------------------------------------------------------
template <bool OUT_BF16>
__global__ void __launch_bounds__(1024)
ligru_recurrent(const __bf16* __restrict__ gh, const __bf16* __restrict__ gz,
                const __bf16* __restrict__ Uh, const __bf16* __restrict__ Uz,
                void* __restrict__ out) {
    constexpr int HP = HID + 8;                 // padded bf16 stride (bank spread)
    constexpr int KH = HID / 2;                 // 256: split-K half
    __shared__ float  hs[BATCH * HID];          // fp32 master copy of h
    __shared__ __bf16 hb[BATCH * HP];           // bf16 copy (WMMA A operand)

    for (int i = threadIdx.x; i < BATCH * HID; i += blockDim.x) hs[i] = 0.0f;
    for (int i = threadIdx.x; i < BATCH * HP;  i += blockDim.x) hb[i] = (__bf16)0.0f;
    __syncthreads();

    const int lane    = threadIdx.x & 31;
    const int wave    = threadIdx.x >> 5;
    const int n0      = wave * 16;
    const int n       = n0 + (lane & 15);
    const int mbase   = (lane >> 4) * 8;
    const int arow    = lane & 15;
    const int khalfA  = (lane >> 4) * 16;
    const int kstartB = (lane >> 4) * 16;

    const __bf16* uhP = Uh + (long)n * HID + kstartB;
    const __bf16* uzP = Uz + (long)n * HID + kstartB;
    const __bf16* hbP = hb + arow * HP + khalfA;

    __bf16* outb = (__bf16*)out;
    float*  outf = (float*)out;

    for (int t = 0; t < T_STEPS; ++t) {
        // Prefetch next step's gate tiles for this wave's columns (16 lines/gate).
        if (t + 1 < T_STEPS) {
            const long gn = (long)(t + 1) * BATCH * HID + (long)(lane & 15) * HID + n0;
            __builtin_prefetch(&gh[gn], 0, 3);
            __builtin_prefetch(&gz[gn], 0, 3);
        }

        v8f ah0 = {}, ah1 = {}, az0 = {}, az1 = {};
#pragma unroll 4
        for (int k0 = 0; k0 < KH; k0 += 32) {
            v16bf a0  = join16(hbP + k0,      hbP + k0 + 8);
            v16bf a1  = join16(hbP + k0 + KH, hbP + k0 + KH + 8);
            v16bf bh0 = join16(uhP + k0,      uhP + k0 + 8);
            v16bf bh1 = join16(uhP + k0 + KH, uhP + k0 + KH + 8);
            v16bf bz0 = join16(uzP + k0,      uzP + k0 + 8);
            v16bf bz1 = join16(uzP + k0 + KH, uzP + k0 + KH + 8);
            ah0 = wmma_bf16(a0, bh0, ah0);
            ah1 = wmma_bf16(a1, bh1, ah1);
            az0 = wmma_bf16(a0, bz0, az0);
            az1 = wmma_bf16(a1, bz1, az1);
        }
        __syncthreads();   // everyone done reading hb before it is overwritten

        const long gbase = (long)t * BATCH * HID;
#pragma unroll
        for (int r = 0; r < 8; ++r) {
            int  m    = mbase + r;
            long gi   = gbase + (long)m * HID + n;
            float whv = (float)gh[gi];
            float wzv = (float)gz[gi];
            float z   = 1.0f / (1.0f + __expf(-(wzv + az0[r] + az1[r])));
            float hc  = whv + ah0[r] + ah1[r];
            hc        = hc > 0.0f ? hc : 0.0f;
            float hp  = hs[m * HID + n];
            float hn  = z * hp + (1.0f - z) * hc;
            hs[m * HID + n] = hn;
            hb[m * HP  + n] = (__bf16)hn;
            if (OUT_BF16) outb[gi] = (__bf16)hn;
            else          outf[gi] = hn;
        }
        __syncthreads();   // h update visible before next step's WMMAs
    }
}

// ---------------------------------------------------------------------------
extern "C" void kernel_launch(void* const* d_in, const int* in_sizes, int n_in,
                              void* d_out, int out_size, void* d_ws, size_t ws_size,
                              hipStream_t stream) {
    (void)in_sizes; (void)n_in; (void)out_size; (void)ws_size;

    const float* x   = (const float*)d_in[0];
    const float* Wh0 = (const float*)d_in[1];
    const float* bh0 = (const float*)d_in[2];
    const float* Wz0 = (const float*)d_in[3];
    const float* bz0 = (const float*)d_in[4];
    const float* Uh0 = (const float*)d_in[5];
    const float* Uz0 = (const float*)d_in[6];
    const float* Wh1 = (const float*)d_in[7];
    const float* bh1 = (const float*)d_in[8];
    const float* Wz1 = (const float*)d_in[9];
    const float* bz1 = (const float*)d_in[10];
    const float* Uh1 = (const float*)d_in[11];
    const float* Uz1 = (const float*)d_in[12];

    char*  ws  = (char*)d_ws;
    size_t off = 0;
    auto alloc = [&](size_t bytes) -> void* {
        void* p = ws + off;
        off += (bytes + 255) & ~(size_t)255;
        return p;
    };

    __bf16* xpad  = (__bf16*)alloc((size_t)ROWS * FEAT_PAD * 2);
    __bf16* wh0p  = (__bf16*)alloc((size_t)HID * FEAT_PAD * 2);
    __bf16* wz0p  = (__bf16*)alloc((size_t)HID * FEAT_PAD * 2);
    __bf16* uh0b  = (__bf16*)alloc((size_t)HID * HID * 2);
    __bf16* uz0b  = (__bf16*)alloc((size_t)HID * HID * 2);
    __bf16* wh1b  = (__bf16*)alloc((size_t)HID * HID * 2);
    __bf16* wz1b  = (__bf16*)alloc((size_t)HID * HID * 2);
    __bf16* uh1b  = (__bf16*)alloc((size_t)HID * HID * 2);
    __bf16* uz1b  = (__bf16*)alloc((size_t)HID * HID * 2);
    __bf16* gh0   = (__bf16*)alloc((size_t)ROWS * HID * 2);
    __bf16* gz0   = (__bf16*)alloc((size_t)ROWS * HID * 2);
    __bf16* gh1   = (__bf16*)alloc((size_t)ROWS * HID * 2);
    __bf16* gz1   = (__bf16*)alloc((size_t)ROWS * HID * 2);
    __bf16* h1bf  = (__bf16*)alloc((size_t)ROWS * HID * 2);

    const int TB = 256;
    auto blocks = [&](long nelem) { return (unsigned)((nelem + TB - 1) / TB); };

    // fp32 -> bf16 conversions (with K padding where needed)
    cvt_pad<<<blocks((long)ROWS * FEAT_PAD), TB, 0, stream>>>(x,   xpad, ROWS, FEAT, FEAT_PAD);
    cvt_pad<<<blocks((long)HID * FEAT_PAD),  TB, 0, stream>>>(Wh0, wh0p, HID,  FEAT, FEAT_PAD);
    cvt_pad<<<blocks((long)HID * FEAT_PAD),  TB, 0, stream>>>(Wz0, wz0p, HID,  FEAT, FEAT_PAD);
    cvt_pad<<<blocks((long)HID * HID), TB, 0, stream>>>(Uh0, uh0b, HID, HID, HID);
    cvt_pad<<<blocks((long)HID * HID), TB, 0, stream>>>(Uz0, uz0b, HID, HID, HID);
    cvt_pad<<<blocks((long)HID * HID), TB, 0, stream>>>(Wh1, wh1b, HID, HID, HID);
    cvt_pad<<<blocks((long)HID * HID), TB, 0, stream>>>(Wz1, wz1b, HID, HID, HID);
    cvt_pad<<<blocks((long)HID * HID), TB, 0, stream>>>(Uh1, uh1b, HID, HID, HID);
    cvt_pad<<<blocks((long)HID * HID), TB, 0, stream>>>(Uz1, uz1b, HID, HID, HID);

    dim3 ggrid(ROWS / 16, HID / 16 / 8);   // (1500, 4)

    // Layer 0: fused gate projections, then sequential recurrence.
    proj_gemm2<<<ggrid, 256, 0, stream>>>(xpad, wh0p, wz0p, bh0, bz0, gh0, gz0, FEAT_PAD);
    ligru_recurrent<true><<<1, 1024, 0, stream>>>(gh0, gz0, uh0b, uz0b, h1bf);

    // Layer 1
    proj_gemm2<<<ggrid, 256, 0, stream>>>(h1bf, wh1b, wz1b, bh1, bz1, gh1, gz1, HID);
    ligru_recurrent<false><<<1, 1024, 0, stream>>>(gh1, gz1, uh1b, uz1b, d_out);
}
